// newHashedLayer_15513421873630
// MI455X (gfx1250) — compile-verified, matrix operand order checked by reading, not run on hardware
//
#include <hip/hip_runtime.h>

// Problem constants (from the reference)
#define FAN_IN  4096
#define FAN_OUT 4096
#define KBUCKET 65536
#define BATCH   8192

// GEMM tiling: block = 128(M) x 256(N), K-slab = 64.
// 8 waves arranged 2(M) x 4(N); each wave computes 64x64 = 4x4 WMMA fragments.
#define BM 128
#define BN 256
#define BK 64
#define LDT 72   // padded LDS row stride in bf16 elements (144B = 36 dwords)

typedef __attribute__((ext_vector_type(16))) __bf16 v16bf;
typedef __attribute__((ext_vector_type(8)))  __bf16 bf16x8;
typedef __attribute__((ext_vector_type(8)))  float  v8f;

static __device__ __forceinline__ unsigned pack_bf16x2(float a, float b) {
  unsigned ua = __builtin_bit_cast(unsigned, a);
  unsigned ub = __builtin_bit_cast(unsigned, b);
  ua += 0x7FFFu + ((ua >> 16) & 1u);      // RNE
  ub += 0x7FFFu + ((ub >> 16) & 1u);
  return (ua >> 16) | (ub & 0xFFFF0000u); // low half = a, high half = b
}

static __device__ __forceinline__ __bf16 f32_to_bf16_rne(float f) {
  unsigned int u = __builtin_bit_cast(unsigned int, f);
  u += 0x7FFFu + ((u >> 16) & 1u);
  unsigned short h = (unsigned short)(u >> 16);
  return __builtin_bit_cast(__bf16, h);
}

static __device__ __forceinline__ v16bf pack16(bf16x8 lo, bf16x8 hi) {
  v16bf r;
#pragma unroll
  for (int i = 0; i < 8; ++i) { r[i] = lo[i]; r[i + 8] = hi[i]; }
  return r;
}

// ---------------------------------------------------------------------------
// Kernel 1: HashedNets expansion. W[o][j] = sign(o,j) * Wc[h(o,j)] as bf16,
// row-major [FAN_OUT][FAN_IN]. Wc (256KB) is cache-resident; the 2x64MB
// index streams are each read exactly once.
// ---------------------------------------------------------------------------
__global__ __launch_bounds__(256) void build_w_bf16(
    const float* __restrict__ Wc,
    const int*   __restrict__ hidx,
    const int*   __restrict__ sbits,
    __bf16*      __restrict__ Wt) {
  size_t base = ((size_t)blockIdx.x * blockDim.x + threadIdx.x) * 8;
  const int4* h4 = reinterpret_cast<const int4*>(hidx + base);
  const int4* s4 = reinterpret_cast<const int4*>(sbits + base);
  int4 h0 = h4[0], h1 = h4[1];
  int4 s0 = s4[0], s1 = s4[1];

  int hv[8] = {h0.x, h0.y, h0.z, h0.w, h1.x, h1.y, h1.z, h1.w};
  int sv[8] = {s0.x, s0.y, s0.z, s0.w, s1.x, s1.y, s1.z, s1.w};

  bf16x8 o;
#pragma unroll
  for (int i = 0; i < 8; ++i) {
    unsigned int u = __builtin_bit_cast(unsigned int, Wc[hv[i]]);
    u ^= ((unsigned int)sv[i]) << 31;          // {0,1} -> {+1,-1} sign flip
    o[i] = f32_to_bf16_rne(__builtin_bit_cast(float, u));
  }
  *reinterpret_cast<bf16x8*>(Wt + base) = o;
}

// ---------------------------------------------------------------------------
// Kernel 2: out = x @ W^T + bias, bf16 WMMA with fp32 accumulation.
// ---------------------------------------------------------------------------
__global__ __launch_bounds__(256) void hashed_gemm(
    const float*  __restrict__ x,     // [BATCH][FAN_IN] f32
    const __bf16* __restrict__ Wt,    // [FAN_OUT][FAN_IN] bf16
    const float*  __restrict__ bias,  // [FAN_OUT]
    float*        __restrict__ out) { // [BATCH][FAN_OUT]
  __shared__ __bf16 sA[BM * LDT];     // rows = m, cols = k   (18.4 KB)
  __shared__ __bf16 sB[BN * LDT];     // rows = n, cols = k   (36.9 KB)

  const int tid  = threadIdx.x;
  const int lane = tid & 31;
  const int wave = tid >> 5;
  const int wm   = wave >> 2;   // 0..1 -> M offset 0 / 64
  const int wn   = wave & 3;    // 0..3 -> N offset 0 / 64 / 128 / 192
  const int lrow = lane & 15;
  const int half = lane >> 4;
  const int nb   = blockIdx.x * BN;
  const int mb   = blockIdx.y * BM;

  v8f acc[4][4];
#pragma unroll
  for (int mi = 0; mi < 4; ++mi)
#pragma unroll
    for (int ni = 0; ni < 4; ++ni)
#pragma unroll
      for (int r = 0; r < 8; ++r) acc[mi][ni][r] = 0.0f;

  for (int kb = 0; kb < FAN_IN; kb += BK) {
    // ---- stage A tile: x fp32 -> packed bf16x2 -> LDS (128 x 64) ----
    {
      const int cg = (tid & 15) * 4;        // k offset within tile
      const int r0 = tid >> 4;              // 0..15
#pragma unroll
      for (int r = 0; r < 8; ++r) {
        const int row = r0 + r * 16;
        float4 v = *reinterpret_cast<const float4*>(
            x + (size_t)(mb + row) * FAN_IN + kb + cg);
        uint2 p;
        p.x = pack_bf16x2(v.x, v.y);
        p.y = pack_bf16x2(v.z, v.w);
        *reinterpret_cast<uint2*>(&sA[row * LDT + cg]) = p;   // ds_store_b64
      }
    }
    // ---- stage B tile: Wt bf16 -> LDS (256 x 64) ----
    {
      const int cg = (tid & 7) * 8;
      const int r0 = tid >> 3;              // 0..31
#pragma unroll
      for (int r = 0; r < 8; ++r) {
        const int row = r0 + r * 32;        // 0..255
        bf16x8 v = *reinterpret_cast<const bf16x8*>(
            Wt + (size_t)(nb + row) * FAN_IN + kb + cg);
        *reinterpret_cast<bf16x8*>(&sB[row * LDT + cg]) = v;
      }
    }
    // prefetch next K-slab while we compute on this one
    if (kb + BK < FAN_IN) {
      __builtin_prefetch(x  + (size_t)(mb + (tid >> 1)) * FAN_IN + kb + BK, 0, 1);
      __builtin_prefetch(Wt + (size_t)(nb + tid) * FAN_IN + kb + BK, 0, 1);
    }
    __syncthreads();

#pragma unroll
    for (int kk = 0; kk < BK; kk += 32) {
      // K sub-offset per the 16-bit A-matrix VGPR layout:
      // lanes 0-15 carry K {0..7,16..23}, lanes 16-31 carry K {8..15,24..31}
      const int kc = kk + half * 8;

      v16bf afrag[4];
#pragma unroll
      for (int mi = 0; mi < 4; ++mi) {
        const __bf16* p = &sA[(wm * 64 + mi * 16 + lrow) * LDT + kc];
        bf16x8 lo = *reinterpret_cast<const bf16x8*>(p);
        bf16x8 hi = *reinterpret_cast<const bf16x8*>(p + 16);
        afrag[mi] = pack16(lo, hi);
      }
      // B loaded once per ni, consumed by 4 WMMAs -> 16 ds_load : 16 wmma
#pragma unroll
      for (int ni = 0; ni < 4; ++ni) {
        const __bf16* p = &sB[(wn * 64 + ni * 16 + lrow) * LDT + kc];
        bf16x8 lo = *reinterpret_cast<const bf16x8*>(p);
        bf16x8 hi = *reinterpret_cast<const bf16x8*>(p + 16);
        v16bf bfrag = pack16(lo, hi);
#pragma unroll
        for (int mi = 0; mi < 4; ++mi)
          acc[mi][ni] = __builtin_amdgcn_wmma_f32_16x16x32_bf16(
              false, afrag[mi], false, bfrag,
              (short)0, acc[mi][ni], false, false);
      }
    }
    __syncthreads();
  }

  // ---- epilogue: bias add + store (C/D layout: VGPR r -> M=r+8*half) ----
#pragma unroll
  for (int ni = 0; ni < 4; ++ni) {
    const int n = nb + wn * 64 + ni * 16 + lrow;
    const float bn = bias[n];
#pragma unroll
    for (int mi = 0; mi < 4; ++mi) {
#pragma unroll
      for (int r = 0; r < 8; ++r) {
        const int m = mb + wm * 64 + mi * 16 + r + 8 * half;
        out[(size_t)m * FAN_OUT + n] = acc[mi][ni][r] + bn;
      }
    }
  }
}

extern "C" void kernel_launch(void* const* d_in, const int* in_sizes, int n_in,
                              void* d_out, int out_size, void* d_ws, size_t ws_size,
                              hipStream_t stream) {
  const float* x     = (const float*)d_in[0];
  const float* Wc    = (const float*)d_in[1];
  const float* bias  = (const float*)d_in[2];
  const int*   hidx  = (const int*)d_in[3];
  const int*   sbits = (const int*)d_in[4];
  float*       out   = (float*)d_out;
  __bf16*      Wt    = (__bf16*)d_ws;   // 4096*4096*2 = 32 MB scratch

  // Phase 1: expand hashed weights to bf16 (once per launch).
  {
    const long total = (long)FAN_OUT * FAN_IN / 8;   // 8 elements / thread
    const int  blocks = (int)(total / 256);          // 8192
    build_w_bf16<<<blocks, 256, 0, stream>>>(Wc, hidx, sbits, Wt);
  }
  // Phase 2: tiled WMMA GEMM.
  {
    dim3 grid(FAN_OUT / BN, BATCH / BM);             // (16, 64)
    hashed_gemm<<<grid, 256, 0, stream>>>(x, Wt, bias, out);
  }
}
